// EntropyGate_4011499454830
// MI455X (gfx1250) — matrix operand order repaired and checked by reading, not compiled, not run on hardware
//
#include <hip/hip_runtime.h>
#include <hip/hip_bf16.h>
#include <math.h>

// ---------------------------------------------------------------------------
// EntropyGate fused pipeline for MI455X (gfx1250, wave32, WMMA).
//   K1: row norms of x             (16384 rows x 2048)
//   K2: windowed softmax entropy   (16384)
//   K2b: pre-swizzle W1/W2 f32 -> bf16 in WMMA B-fragment order (one-time)
//   K3: GEMM1 [y_ssm|y_attn] @ W1[0:4096] (+ H*W1[4096] + b1) -> SiLU -> h(bf16)
//   K4: GEMM2 h @ W2 (+ b2) -> sigmoid -> blend(y_ssm,y_attn) -> out(f32)
//
// GEMM: 128x128 block tile, TK=32, 8 waves/block; each wave = 16(M) x 128(N)
// via 8x v_wmma_f32_16x16x32_bf16 per K-step.
//  - A fragments from global, software-pipelined one K-step ahead in registers
//  - B tiles: pre-swizzled bf16 -> LDS via global_load_async_to_lds_b128,
//    TRIPLE buffered; per-iteration wait is asynccnt<=2 so we only ever wait
//    on the tile issued two K-steps ago (ASYNCcnt completes in order).
//  - sched_barrier splits the 16x ds_load_b128 clause from the WMMA burst so
//    s_wait_dscnt staggers and LDS latency hides under matrix ops.
// ---------------------------------------------------------------------------

typedef __attribute__((ext_vector_type(16))) __bf16 v16bf;
typedef __attribute__((ext_vector_type(8)))  __bf16 v8bf;
typedef __attribute__((ext_vector_type(2)))  __bf16 v2bf;
typedef __attribute__((ext_vector_type(8)))  float  v8f;

#define EG_D   2048
#define EG_S   4096
#define EG_TM  128
#define EG_TN  128
#define EG_TK  32
#define EG_NT  (EG_D >> 4)         // 128 n-tiles across full N
#define EG_FRAG 512                // bf16 elems per (k-tile, n-tile) fragment image

// ---------------- CDNA5 async global->LDS helpers ----------------
__device__ inline void eg_async_ld16(unsigned lds_addr, const void* gsrc) {
    asm volatile("global_load_async_to_lds_b128 %0, %1, off"
                 :: "v"(lds_addr), "v"(gsrc)
                 : "memory");
}

__device__ inline void eg_wait_async0() {
#if __has_builtin(__builtin_amdgcn_s_wait_asynccnt)
    __builtin_amdgcn_s_wait_asynccnt(0);
#else
    asm volatile("s_wait_asynccnt 0x0" ::: "memory");
#endif
}

__device__ inline void eg_wait_async2() {
#if __has_builtin(__builtin_amdgcn_s_wait_asynccnt)
    __builtin_amdgcn_s_wait_asynccnt(2);
#else
    asm volatile("s_wait_asynccnt 0x2" ::: "memory");
#endif
}

__device__ inline void eg_sched_split() {
#if __has_builtin(__builtin_amdgcn_sched_barrier)
    __builtin_amdgcn_sched_barrier(0);
#endif
}

// Copy one pre-swizzled 8KB B tile (contiguous) global -> LDS: 32B per thread.
__device__ inline void eg_async_copy_btile(__bf16* dst, const __bf16* src, int tid) {
    const __bf16* s = src + tid * 16;
    __bf16* d = dst + tid * 16;
    // LDS aperture: low 32 bits of the generic address are the LDS byte offset.
    unsigned la = (unsigned)(uintptr_t)d;
    eg_async_ld16(la,       (const void*)s);
    eg_async_ld16(la + 16u, (const void*)(s + 8));
}

// ---------------- K1: row L2 norms ----------------
__global__ __launch_bounds__(256) void eg_norm_kernel(const float* __restrict__ x,
                                                      float* __restrict__ norms) {
    __shared__ float red[256];
    const int row = blockIdx.x;
    const float* xr = x + (size_t)row * EG_D;
    float s = 0.f;
    for (int i = threadIdx.x; i < EG_D; i += 256) {
        float v = xr[i];
        s += v * v;
    }
    red[threadIdx.x] = s;
    __syncthreads();
    for (int off = 128; off > 0; off >>= 1) {
        if (threadIdx.x < off) red[threadIdx.x] += red[threadIdx.x + off];
        __syncthreads();
    }
    if (threadIdx.x == 0) norms[row] = sqrtf(red[0]);
}

// ---------------- K2: sliding-window softmax entropy ----------------
__global__ __launch_bounds__(256) void eg_entropy_kernel(const float* __restrict__ norms,
                                                         float* __restrict__ Hout,
                                                         int M) {
    int r = blockIdx.x * blockDim.x + threadIdx.x;
    if (r >= M) return;
    int b = r / EG_S;
    int s = r - b * EG_S;
    float w[8];
    bool  valid[8];
    float mx = -INFINITY;
    #pragma unroll
    for (int j = 0; j < 8; ++j) {
        int idx = s + j - 7;                 // offsets -7..0
        valid[j] = (idx >= 0);
        int ic = idx < 0 ? 0 : idx;
        float v = valid[j] ? norms[(size_t)b * EG_S + ic] : -INFINITY;
        w[j] = v;
        mx = fmaxf(mx, v);
    }
    float den = 0.f;
    float e[8];
    #pragma unroll
    for (int j = 0; j < 8; ++j) {
        e[j] = valid[j] ? __expf(w[j] - mx) : 0.f;
        den += e[j];
    }
    float inv = 1.f / den;
    float H = 0.f;
    #pragma unroll
    for (int j = 0; j < 8; ++j) {
        float p = e[j] * inv;
        H -= p * log2f(p + 1e-9f);
    }
    Hout[r] = H;
}

// ---------------- K2b: weight pre-swizzle (f32 KxN -> bf16 fragment order) ----
// Output order: [kt][nt][lane][j], j = CDNA5 table K order (16 elems);
// lane = (n&15) + ((k&8)<<1), j = (k&7) + ((k&16)>>1).  One thread per (even k, n): packs a b32.
__global__ __launch_bounds__(256) void eg_wswz_kernel(const float* __restrict__ W,
                                                      __bf16* __restrict__ Wc,
                                                      int K) {
    int tid = blockIdx.x * 256 + threadIdx.x;
    if (tid >= (K >> 1) * EG_D) return;
    int n  = tid & (EG_D - 1);
    int k  = (tid >> 11) << 1;            // even k  (log2(EG_D) = 11)
    float a = W[(size_t)k * EG_D + n];
    float b = W[(size_t)(k + 1) * EG_D + n];
    int kt = k >> 5, kl = k & 31;
    int nt = n >> 4, nn = n & 15;
    int lane = nn + ((kl & 8) << 1);
    int j = (kl & 7) + ((kl & 16) >> 1);   // even
    size_t idx = (((size_t)(kt * EG_NT + nt) << 5) + lane) * 16 + j;
    v2bf p;
    p[0] = (__bf16)a;
    p[1] = (__bf16)b;
    *(v2bf*)(Wc + idx) = p;                // 4B-aligned (j even)
}

// ---------------- fragment helpers ----------------
// A 16x32 bf16 frag (CDNA5 layout: lane<16 -> K {0-7,16-23}; lane>=16 ->
// {8-15,24-31}).  Raw f32 quads held pipelined in registers, packed here.
__device__ inline void eg_load_a_f32(const float* __restrict__ rowk, int khalf,
                                     float4 f[4]) {
    f[0] = *(const float4*)(rowk + khalf);
    f[1] = *(const float4*)(rowk + khalf + 4);
    f[2] = *(const float4*)(rowk + khalf + 16);
    f[3] = *(const float4*)(rowk + khalf + 20);
}

__device__ inline v16bf eg_pack_a(const float4 f[4]) {
    v16bf a;
    a[0]  = (__bf16)f[0].x; a[1]  = (__bf16)f[0].y; a[2]  = (__bf16)f[0].z; a[3]  = (__bf16)f[0].w;
    a[4]  = (__bf16)f[1].x; a[5]  = (__bf16)f[1].y; a[6]  = (__bf16)f[1].z; a[7]  = (__bf16)f[1].w;
    a[8]  = (__bf16)f[2].x; a[9]  = (__bf16)f[2].y; a[10] = (__bf16)f[2].z; a[11] = (__bf16)f[2].w;
    a[12] = (__bf16)f[3].x; a[13] = (__bf16)f[3].y; a[14] = (__bf16)f[3].z; a[15] = (__bf16)f[3].w;
    return a;
}

__device__ inline void eg_load_a_bf16(const __bf16* __restrict__ rowk, int khalf,
                                      v8bf h2[2]) {
    h2[0] = *(const v8bf*)(rowk + khalf);
    h2[1] = *(const v8bf*)(rowk + khalf + 16);
}

__device__ inline v16bf eg_pack_a_bf16(const v8bf h2[2]) {
    return __builtin_shufflevector(h2[0], h2[1], 0, 1, 2, 3, 4, 5, 6, 7,
                                   8, 9, 10, 11, 12, 13, 14, 15);
}

// ---------------- K3: GEMM1 + rank-1 H term + bias + SiLU -> h(bf16) --------
__global__ __launch_bounds__(256) void eg_gemm1_kernel(const float* __restrict__ ys,
                                                       const float* __restrict__ ya,
                                                       const __bf16* __restrict__ W1c,
                                                       const float* __restrict__ W1,
                                                       const float* __restrict__ b1,
                                                       const float* __restrict__ H,
                                                       __bf16* __restrict__ hout) {
    __shared__ alignas(32) __bf16 Bs[3][EG_TK * EG_TN];   // 3 x 8KB, fragment order

    const int tid  = threadIdx.x;
    const int wave = tid >> 5;
    const int lane = tid & 31;
    const int m0   = blockIdx.y * EG_TM;
    const int n0   = blockIdx.x * EG_TN;

    const int mrow  = m0 + wave * 16 + (lane & 15);
    const int khalf = (lane >> 4) * 8;
    const float* ysrow = ys + (size_t)mrow * EG_D;
    const float* yarow = ya + (size_t)mrow * EG_D;

    const size_t bstride = (size_t)EG_NT * EG_FRAG;           // elems per k-tile
    const __bf16* bsrc = W1c + (size_t)(n0 >> 4) * EG_FRAG;   // + kt*bstride

    v8f acc[8] = {};

    // Prologue: async-stage B tiles 0 and 1; pipeline A floats for kt=0.
    eg_async_copy_btile(Bs[0], bsrc, tid);
    eg_async_copy_btile(Bs[1], bsrc + bstride, tid);
    float4 fA[4];
    eg_load_a_f32(ysrow, khalf, fA);
    eg_wait_async2();          // in-order: tile 0 is complete
    __syncthreads();

    const int KT = (2 * EG_D) / EG_TK;   // 128
    int cur = 0;
    for (int kt = 0; kt < KT; ++kt) {
        const __bf16* curp = &Bs[cur][0];
        v16bf af = eg_pack_a(fA);
        if (kt + 1 < KT) {
            // Pipeline next A fragment's raw floats (latency hidden by MMA).
            const int kk = (kt + 1) * EG_TK;   // never straddles the 2048 seam
            const float* rowk = (kk < EG_D) ? (ysrow + kk) : (yarow + (kk - EG_D));
            eg_load_a_f32(rowk, khalf, fA);
        }
        if (kt + 2 < KT) {
            // Async-stage tile kt+2 two steps ahead (triple buffer).
            int b2i = cur + 2; if (b2i >= 3) b2i -= 3;
            eg_async_copy_btile(Bs[b2i], bsrc + (size_t)(kt + 2) * bstride, tid);
        }
        // Load all 8 B fragments, then split scheduling so the WMMA burst
        // overlaps the staggered ds waits.
        v16bf bfr[8];
        #pragma unroll
        for (int nt = 0; nt < 8; ++nt)
            bfr[nt] = *(const v16bf*)(curp + (((nt << 5) + lane) << 4));
        eg_sched_split();
        #pragma unroll
        for (int nt = 0; nt < 8; ++nt)
            acc[nt] = __builtin_amdgcn_wmma_f32_16x16x32_bf16(
                false, af, false, bfr[nt], (short)0, acc[nt], false, false);
        eg_wait_async2();      // only waits on the tile issued 2 steps ago
        __syncthreads();
        cur = (cur + 1 == 3) ? 0 : cur + 1;
    }

    // Epilogue: + H[r]*W1[4096,:] + b1, SiLU, store bf16
    const int n_lane = lane & 15;
    const int m_half = (lane >> 4) * 8;
    const float* W1last = W1 + (size_t)(2 * EG_D) * EG_D;
    #pragma unroll
    for (int nt = 0; nt < 8; ++nt) {
        const int gn = n0 + nt * 16 + n_lane;
        const float wl = W1last[gn];
        const float bb = b1[gn];
        #pragma unroll
        for (int v = 0; v < 8; ++v) {
            const int gm = m0 + wave * 16 + m_half + v;
            float val = acc[nt][v] + H[gm] * wl + bb;
            float sil = val / (1.f + __expf(-val));
            hout[(size_t)gm * EG_D + gn] = (__bf16)sil;
        }
    }
}

// ---------------- K4: GEMM2 + bias + sigmoid gate blend -> out(f32) ---------
__global__ __launch_bounds__(256) void eg_gemm2_kernel(const __bf16* __restrict__ h,
                                                       const __bf16* __restrict__ W2c,
                                                       const float* __restrict__ b2,
                                                       const float* __restrict__ ys,
                                                       const float* __restrict__ ya,
                                                       float* __restrict__ out) {
    __shared__ alignas(32) __bf16 Bs[3][EG_TK * EG_TN];

    const int tid  = threadIdx.x;
    const int wave = tid >> 5;
    const int lane = tid & 31;
    const int m0   = blockIdx.y * EG_TM;
    const int n0   = blockIdx.x * EG_TN;

    const int mrow  = m0 + wave * 16 + (lane & 15);
    const int khalf = (lane >> 4) * 8;
    const __bf16* hrow = h + (size_t)mrow * EG_D;

    const size_t bstride = (size_t)EG_NT * EG_FRAG;
    const __bf16* bsrc = W2c + (size_t)(n0 >> 4) * EG_FRAG;

    v8f acc[8] = {};

    eg_async_copy_btile(Bs[0], bsrc, tid);
    eg_async_copy_btile(Bs[1], bsrc + bstride, tid);
    v8bf hA[2];
    eg_load_a_bf16(hrow, khalf, hA);
    eg_wait_async2();
    __syncthreads();

    const int KT = EG_D / EG_TK;   // 64
    int cur = 0;
    for (int kt = 0; kt < KT; ++kt) {
        const __bf16* curp = &Bs[cur][0];
        v16bf af = eg_pack_a_bf16(hA);
        if (kt + 1 < KT)
            eg_load_a_bf16(hrow + (kt + 1) * EG_TK, khalf, hA);
        if (kt + 2 < KT) {
            int b2i = cur + 2; if (b2i >= 3) b2i -= 3;
            eg_async_copy_btile(Bs[b2i], bsrc + (size_t)(kt + 2) * bstride, tid);
        }
        v16bf bfr[8];
        #pragma unroll
        for (int nt = 0; nt < 8; ++nt)
            bfr[nt] = *(const v16bf*)(curp + (((nt << 5) + lane) << 4));
        eg_sched_split();
        #pragma unroll
        for (int nt = 0; nt < 8; ++nt)
            acc[nt] = __builtin_amdgcn_wmma_f32_16x16x32_bf16(
                false, af, false, bfr[nt], (short)0, acc[nt], false, false);
        eg_wait_async2();
        __syncthreads();
        cur = (cur + 1 == 3) ? 0 : cur + 1;
    }

    const int n_lane = lane & 15;
    const int m_half = (lane >> 4) * 8;
    #pragma unroll
    for (int nt = 0; nt < 8; ++nt) {
        const int gn = n0 + nt * 16 + n_lane;
        const float bb = b2[gn];
        #pragma unroll
        for (int v = 0; v < 8; ++v) {
            const int gm = m0 + wave * 16 + m_half + v;
            const size_t idx = (size_t)gm * EG_D + gn;
            float g = 1.f / (1.f + __expf(-(acc[nt][v] + bb)));
            out[idx] = g * ys[idx] + (1.f - g) * ya[idx];
        }
    }
}

// ---------------- Host-side launcher ----------------
extern "C" void kernel_launch(void* const* d_in, const int* in_sizes, int n_in,
                              void* d_out, int out_size, void* d_ws, size_t ws_size,
                              hipStream_t stream) {
    const float* ys = (const float*)d_in[0];
    const float* ya = (const float*)d_in[1];
    const float* x  = (const float*)d_in[2];
    const float* W1 = (const float*)d_in[3];
    const float* b1 = (const float*)d_in[4];
    const float* W2 = (const float*)d_in[5];
    const float* b2 = (const float*)d_in[6];
    float* out = (float*)d_out;

    const int M = in_sizes[0] / EG_D;  // B*S = 16384

    char* ws = (char*)d_ws;
    size_t off = 0;
    float* norms = (float*)(ws + off);  off += (size_t)M * sizeof(float);
    float* Hbuf  = (float*)(ws + off);  off += (size_t)M * sizeof(float);
    __bf16* hbuf = (__bf16*)(ws + off); off += (size_t)M * EG_D * sizeof(__bf16);
    __bf16* W1c  = (__bf16*)(ws + off); off += (size_t)2 * EG_D * EG_D * sizeof(__bf16);
    __bf16* W2c  = (__bf16*)(ws + off);

    eg_norm_kernel<<<M, 256, 0, stream>>>(x, norms);
    eg_entropy_kernel<<<(M + 255) / 256, 256, 0, stream>>>(norms, Hbuf, M);

    const int t1 = EG_D * EG_D;        // (4096/2)*2048
    eg_wswz_kernel<<<(t1 + 255) / 256, 256, 0, stream>>>(W1, W1c, 2 * EG_D);
    const int t2 = (EG_D / 2) * EG_D;
    eg_wswz_kernel<<<(t2 + 255) / 256, 256, 0, stream>>>(W2, W2c, EG_D);

    dim3 grid(EG_D / EG_TN, M / EG_TM);  // (16, 128)
    eg_gemm1_kernel<<<grid, 256, 0, stream>>>(ys, ya, W1c, W1, b1, Hbuf, hbuf);
    eg_gemm2_kernel<<<grid, 256, 0, stream>>>(hbuf, W2c, b2, ys, ya, out);
}
